// PseudoTokenGridEncoderThroughTime_86251533238895
// MI455X (gfx1250) — matrix operand order repaired and checked by reading, not compiled, not run on hardware
//
#include <hip/hip_runtime.h>
#include <hip/hip_bf16.h>

// ---------------------------------------------------------------------------
// PseudoTokenGridEncoderThroughTime for MI455X (gfx1250, wave32, WMMA).
//
// Pipeline:
//   K0: zero per-cell counters (ws)
//   K1: emit x_grid output (trivial)
//   K2: assign each point to its nearest grid cell (atomicAdd slot alloc)
//   Kp: pack Wq/Wk/Wv/Wo -> bf16 transposed [N][K] (B operands, loaded b128)
//   K3: WMMA GEMMs: Lq = lat@Wq; (Lk,Lv) = lat@(Wk,Wv) fused;
//                   (Pk,Pv) = z@(Wk,Wv) fused (reads z once)
//   K4: per-cell 8-head attention (1 wave per head, head_dim 32 == wave32)
//   K5: WMMA GEMM: z_grid = attn_out @ Wo -> d_out (after x_grid)
//
// Workspace layout (bytes):            offset        size
//   counts   int[4*8192]                0             131072
//   cellpts  int[4*8192*15]             131072        1966080
//   Lq       f32[1024*256]              2097152       1048576
//   Lk       f32[1024*256]              3145728       1048576
//   Lv       f32[1024*256]              4194304       1048576
//   Pk       f32[65536*256]             5242880       67108864
//   Pv       f32[65536*256]             72351744      67108864
//   attn_out f32[32768*256]             139460608     33554432
//   WtQ/K/V/O bf16[256*256] x4          173015040     524288
//   total ~173.5 MB
// ---------------------------------------------------------------------------

typedef __attribute__((ext_vector_type(16))) __bf16 v16bf;
typedef __attribute__((ext_vector_type(8)))  __bf16 v8bf;
typedef __attribute__((ext_vector_type(8)))  float  v8f;
typedef __attribute__((ext_vector_type(4)))  float  v4f;

#define EMBED   256
#define HEADS   8
#define HDIM    32
#define TGRID   8
#define NG      32
#define SCELLS  (TGRID * NG * NG)   /* 8192 */
#define MBATCH  4
#define NPTS    16384
#define PSLOT   16                  /* 15 point slots + 1 latent slot */

static constexpr size_t OFF_COUNTS = 0;
static constexpr size_t OFF_CELLPT = 131072;
static constexpr size_t OFF_LQ     = 2097152;
static constexpr size_t OFF_LK     = 3145728;
static constexpr size_t OFF_LV     = 4194304;
static constexpr size_t OFF_PK     = 5242880;
static constexpr size_t OFF_PV     = 72351744;
static constexpr size_t OFF_ATTN   = 139460608;
static constexpr size_t OFF_WTQ    = 173015040;
static constexpr size_t OFF_WTK    = OFF_WTQ + 131072;
static constexpr size_t OFF_WTV    = OFF_WTK + 131072;
static constexpr size_t OFF_WTO    = OFF_WTV + 131072;

static constexpr int XGRID_ELEMS = MBATCH * TGRID * NG * NG * 3;  // 98304

// ---------------------------------------------------------------- K0: zero
__global__ void zero_counts_kernel(int* __restrict__ counts) {
    int gid = blockIdx.x * blockDim.x + threadIdx.x;
    if (gid < MBATCH * SCELLS) counts[gid] = 0;
}

// ---------------------------------------------------------------- K1: x_grid
__global__ void xgrid_kernel(const float* __restrict__ time_grid,
                             const float* __restrict__ grid,
                             float* __restrict__ out) {
    int gid = blockIdx.x * blockDim.x + threadIdx.x;        // [0, m*t*1024)
    if (gid >= MBATCH * TGRID * NG * NG) return;
    int mi  = gid >> 13;
    int rem = gid & (TGRID * NG * NG - 1);
    int ti  = rem >> 10;
    int sp  = rem & 1023;
    float* o = out + (size_t)gid * 3;
    o[0] = time_grid[mi * TGRID + ti];
    o[1] = grid[sp * 2 + 0];
    o[2] = grid[sp * 2 + 1];
}

// ---------------------------------------------------------------- K2: assign
__global__ void assign_kernel(const float* __restrict__ x,
                              const float* __restrict__ time_grid,
                              const float* __restrict__ grid,
                              int* __restrict__ counts,
                              int* __restrict__ cellpts) {
    int gid = blockIdx.x * blockDim.x + threadIdx.x;        // [0, m*n)
    if (gid >= MBATCH * NPTS) return;
    int mi = gid >> 14;
    int pt = gid & (NPTS - 1);

    float xt = x[(size_t)gid * 3 + 0];
    float x0 = x[(size_t)gid * 3 + 1];
    float x1 = x[(size_t)gid * 3 + 2];

    float tmin = time_grid[mi * TGRID + 0];
    float tmax = time_grid[mi * TGRID + TGRID - 1];
    float g0min = grid[0];
    float g1min = grid[1];
    float g0max = grid[((NG - 1) * NG + (NG - 1)) * 2 + 0];
    float g1max = grid[((NG - 1) * NG + (NG - 1)) * 2 + 1];

    float spt = (tmax - tmin) / (float)(TGRID - 1);
    float sp0 = (g0max - g0min) / (float)(NG - 1);
    float sp1 = (g1max - g1min) / (float)(NG - 1);

    float ft = floorf((xt - tmin + 0.5f * spt) / spt);
    float f0 = floorf((x0 - g0min + 0.5f * sp0) / sp0);
    float f1 = floorf((x1 - g1min + 0.5f * sp1) / sp1);
    ft = fminf(fmaxf(ft, 0.0f), (float)(TGRID - 1));
    f0 = fminf(fmaxf(f0, 0.0f), (float)(NG - 1));
    f1 = fminf(fmaxf(f1, 0.0f), (float)(NG - 1));

    int cell = (int)ft * (NG * NG) + (int)f0 * NG + (int)f1;
    int g = mi * SCELLS + cell;
    int slot = atomicAdd(&counts[g], 1);
    if (slot < PSLOT - 1) cellpts[(size_t)g * (PSLOT - 1) + slot] = pt;
}

// ------------------------------------------------- Kp: pack W -> bf16 Wt[N][K]
__global__ void pack_wt_kernel(const float* __restrict__ W,
                               __bf16* __restrict__ Wt) {
    int gid = blockIdx.x * blockDim.x + threadIdx.x;        // [0, 65536)
    if (gid >= EMBED * EMBED) return;
    int n = gid >> 8;
    int k = gid & 255;
    Wt[(size_t)n * EMBED + k] = (__bf16)W[(size_t)k * EMBED + n];
}

// ------------------------------------------------- WMMA fragment helpers
// A fragment (16x32 bf16): lane-half holds contiguous K block {kb..kb+15},
// kb = 16*(lane>=16).  p points at arow + k0 + kb -> one contiguous 16-float
// run (64B aligned -> 4x global_load_b128).
__device__ __forceinline__ v16bf load_afrag_f32(const float* __restrict__ p) {
    const v4f* q = (const v4f*)p;
    v4f a0 = q[0], a1 = q[1], a2 = q[2], a3 = q[3];
    v16bf r;
#pragma unroll
    for (int e = 0; e < 4; ++e) {
        r[e]      = (__bf16)a0[e];
        r[e + 4]  = (__bf16)a1[e];
        r[e + 8]  = (__bf16)a2[e];
        r[e + 12] = (__bf16)a3[e];
    }
    return r;
}

// B fragment (32x16 bf16): lane-half holds contiguous K block of 16 at the
// lane's column.  Wt is [N][K] bf16 -> 32B contiguous = 2x global_load_b128.
__device__ __forceinline__ v16bf load_bfrag_bf16(const __bf16* __restrict__ p) {
    const v8bf* q = (const v8bf*)p;
    v8bf lo = q[0], hi = q[1];
    return __builtin_shufflevector(lo, hi, 0, 1, 2, 3, 4, 5, 6, 7,
                                           8, 9, 10, 11, 12, 13, 14, 15);
}

// ------------------------------------------------- K3a: C = A @ W (1 output)
// Block = 256 threads = 8 waves; block tile 32x256; wave tile 16x64.
__global__ __launch_bounds__(256) void gemm256_wmma_1(
        const float*  __restrict__ A,
        const __bf16* __restrict__ Wt,      // [N][K] bf16
        float* __restrict__ C) {
    const int lane = threadIdx.x & 31;
    const int wave = threadIdx.x >> 5;
    const int row0 = blockIdx.x * 32 + (wave >> 2) * 16;
    const int col0 = (wave & 3) * 64;
    const int nm   = lane & 15;
    const int kbA  = (lane & 16) ? 16 : 0;
    const int kbB  = (lane & 16) ? 16 : 0;

    v8f acc[4] = {v8f{}, v8f{}, v8f{}, v8f{}};
    const float* __restrict__ arow = A + (size_t)(row0 + nm) * EMBED + kbA;

    for (int k0 = 0; k0 < EMBED; k0 += 32) {
        v16bf afrag = load_afrag_f32(arow + k0);
        if (k0 + 32 < EMBED) __builtin_prefetch(arow + k0 + 32, 0, 0);
#pragma unroll
        for (int f = 0; f < 4; ++f) {
            const int nc = col0 + f * 16 + nm;
            v16bf b = load_bfrag_bf16(Wt + (size_t)nc * EMBED + k0 + kbB);
            acc[f] = __builtin_amdgcn_wmma_f32_16x16x32_bf16(
                false, afrag, false, b, (short)0, acc[f], false, false);
        }
    }

    const int rbase = row0 + ((lane & 16) ? 8 : 0);
#pragma unroll
    for (int f = 0; f < 4; ++f) {
        const int nc = col0 + f * 16 + nm;
#pragma unroll
        for (int v = 0; v < 8; ++v)
            C[(size_t)(rbase + v) * EMBED + nc] = acc[f][v];
    }
}

// ------------------------------------------------- K3b: dual-output GEMM
// C1 = A @ W1, C2 = A @ W2 — A fragments loaded ONCE (halves z traffic).
__global__ __launch_bounds__(256) void gemm256_wmma_2(
        const float*  __restrict__ A,
        const __bf16* __restrict__ Wt1,
        const __bf16* __restrict__ Wt2,
        float* __restrict__ C1,
        float* __restrict__ C2) {
    const int lane = threadIdx.x & 31;
    const int wave = threadIdx.x >> 5;
    const int row0 = blockIdx.x * 32 + (wave >> 2) * 16;
    const int col0 = (wave & 3) * 64;
    const int nm   = lane & 15;
    const int kbA  = (lane & 16) ? 16 : 0;
    const int kbB  = (lane & 16) ? 16 : 0;

    v8f acc1[4] = {v8f{}, v8f{}, v8f{}, v8f{}};
    v8f acc2[4] = {v8f{}, v8f{}, v8f{}, v8f{}};
    const float* __restrict__ arow = A + (size_t)(row0 + nm) * EMBED + kbA;

    for (int k0 = 0; k0 < EMBED; k0 += 32) {
        v16bf afrag = load_afrag_f32(arow + k0);
        if (k0 + 32 < EMBED) __builtin_prefetch(arow + k0 + 32, 0, 0);
#pragma unroll
        for (int f = 0; f < 4; ++f) {
            const int nc = col0 + f * 16 + nm;
            v16bf b1 = load_bfrag_bf16(Wt1 + (size_t)nc * EMBED + k0 + kbB);
            acc1[f] = __builtin_amdgcn_wmma_f32_16x16x32_bf16(
                false, afrag, false, b1, (short)0, acc1[f], false, false);
            v16bf b2 = load_bfrag_bf16(Wt2 + (size_t)nc * EMBED + k0 + kbB);
            acc2[f] = __builtin_amdgcn_wmma_f32_16x16x32_bf16(
                false, afrag, false, b2, (short)0, acc2[f], false, false);
        }
    }

    const int rbase = row0 + ((lane & 16) ? 8 : 0);
#pragma unroll
    for (int f = 0; f < 4; ++f) {
        const int nc = col0 + f * 16 + nm;
#pragma unroll
        for (int v = 0; v < 8; ++v) {
            C1[(size_t)(rbase + v) * EMBED + nc] = acc1[f][v];
            C2[(size_t)(rbase + v) * EMBED + nc] = acc2[f][v];
        }
    }
}

// ---------------------------------------------------------------- K4: attn
// One block per cell; wave w handles head w (HDIM 32 == wave32 lanes).
__global__ __launch_bounds__(256) void attn_kernel(
        const int*   __restrict__ counts,
        const int*   __restrict__ cellpts,
        const float* __restrict__ Lq,
        const float* __restrict__ Lk,
        const float* __restrict__ Lv,
        const float* __restrict__ Pk,
        const float* __restrict__ Pv,
        float* __restrict__ out) {
    const int g    = blockIdx.x;             // [0, m*S)
    const int mi   = g >> 13;
    const int s    = g & (SCELLS - 1);
    const int lat  = s & (NG * NG - 1);
    const int lane = threadIdx.x & 31;
    const int head = threadIdx.x >> 5;
    const int d    = head * HDIM + lane;

    int cnt = counts[g];
    if (cnt > PSLOT - 1) cnt = PSLOT - 1;

    const float q = Lq[(size_t)lat * EMBED + d];
    const float scale = 0.17677669529663687f;   // 1/sqrt(32)
    float logits[PSLOT];

    for (int p = 0; p < cnt; ++p) {
        const int pidx = cellpts[(size_t)g * (PSLOT - 1) + p];
        float prod = q * Pk[((size_t)mi * NPTS + pidx) * EMBED + d];
#pragma unroll
        for (int off = 16; off; off >>= 1) prod += __shfl_xor(prod, off, 32);
        logits[p] = prod * scale;
    }
    {
        float prod = q * Lk[(size_t)lat * EMBED + d];
#pragma unroll
        for (int off = 16; off; off >>= 1) prod += __shfl_xor(prod, off, 32);
        logits[cnt] = prod * scale;
    }

    float mx = logits[cnt];
    for (int p = 0; p < cnt; ++p) mx = fmaxf(mx, logits[p]);
    float denom = 0.0f;
    for (int p = 0; p <= cnt; ++p) { logits[p] = expf(logits[p] - mx); denom += logits[p]; }
    const float inv = 1.0f / denom;

    float o = 0.0f;
    for (int p = 0; p < cnt; ++p) {
        const int pidx = cellpts[(size_t)g * (PSLOT - 1) + p];
        o += logits[p] * inv * Pv[((size_t)mi * NPTS + pidx) * EMBED + d];
    }
    o += logits[cnt] * inv * Lv[(size_t)lat * EMBED + d];

    out[(size_t)g * EMBED + d] = o;
}

// ---------------------------------------------------------------------------
extern "C" void kernel_launch(void* const* d_in, const int* in_sizes, int n_in,
                              void* d_out, int out_size, void* d_ws, size_t ws_size,
                              hipStream_t stream) {
    const float* x   = (const float*)d_in[0];
    const float* z   = (const float*)d_in[1];
    const float* tgr = (const float*)d_in[2];
    const float* lat = (const float*)d_in[3];
    const float* grd = (const float*)d_in[4];
    const float* Wq  = (const float*)d_in[5];
    const float* Wk  = (const float*)d_in[6];
    const float* Wv  = (const float*)d_in[7];
    const float* Wo  = (const float*)d_in[8];
    float* out = (float*)d_out;

    char* ws = (char*)d_ws;
    int*    counts  = (int*)   (ws + OFF_COUNTS);
    int*    cellpts = (int*)   (ws + OFF_CELLPT);
    float*  LqB     = (float*) (ws + OFF_LQ);
    float*  LkB     = (float*) (ws + OFF_LK);
    float*  LvB     = (float*) (ws + OFF_LV);
    float*  PkB     = (float*) (ws + OFF_PK);
    float*  PvB     = (float*) (ws + OFF_PV);
    float*  attnO   = (float*) (ws + OFF_ATTN);
    __bf16* WtQ     = (__bf16*)(ws + OFF_WTQ);
    __bf16* WtK     = (__bf16*)(ws + OFF_WTK);
    __bf16* WtV     = (__bf16*)(ws + OFF_WTV);
    __bf16* WtO     = (__bf16*)(ws + OFF_WTO);

    // K0: zero per-cell counters
    zero_counts_kernel<<<(MBATCH * SCELLS + 255) / 256, 256, 0, stream>>>(counts);

    // K1: x_grid output
    xgrid_kernel<<<(MBATCH * TGRID * NG * NG + 255) / 256, 256, 0, stream>>>(
        tgr, grd, out);

    // K2: point -> cell assignment
    assign_kernel<<<(MBATCH * NPTS + 255) / 256, 256, 0, stream>>>(
        x, tgr, grd, counts, cellpts);

    // Kp: pack weights to bf16 transposed
    pack_wt_kernel<<<(EMBED * EMBED) / 256, 256, 0, stream>>>(Wq, WtQ);
    pack_wt_kernel<<<(EMBED * EMBED) / 256, 256, 0, stream>>>(Wk, WtK);
    pack_wt_kernel<<<(EMBED * EMBED) / 256, 256, 0, stream>>>(Wv, WtV);
    pack_wt_kernel<<<(EMBED * EMBED) / 256, 256, 0, stream>>>(Wo, WtO);

    // K3: projections (WMMA bf16, f32 accumulate)
    gemm256_wmma_1<<<(NG * NG) / 32, 256, 0, stream>>>(lat, WtQ, LqB);
    gemm256_wmma_2<<<(NG * NG) / 32, 256, 0, stream>>>(lat, WtK, WtV, LkB, LvB);
    gemm256_wmma_2<<<(MBATCH * NPTS) / 32, 256, 0, stream>>>(z, WtK, WtV, PkB, PvB);

    // K4: per-cell masked MHA (1 query/cell)
    attn_kernel<<<MBATCH * SCELLS, 256, 0, stream>>>(
        counts, cellpts, LqB, LkB, LvB, PkB, PvB, attnO);

    // K5: output projection -> z_grid portion of d_out
    gemm256_wmma_1<<<(MBATCH * SCELLS) / 32, 256, 0, stream>>>(
        attnO, WtO, out + XGRID_ELEMS);
}
